// APPNP_5634997092477
// MI455X (gfx1250) — matrix operand order repaired and checked by reading, not compiled, not run on hardware
//
#include <hip/hip_runtime.h>

typedef __attribute__((ext_vector_type(16))) __bf16 v16bf;
typedef __attribute__((ext_vector_type(8)))  __bf16 v8bf;
typedef __attribute__((ext_vector_type(8)))  float  v8f;

#define NN 100000
#define NE 3200000
#define HID 512
#define CLS 64
#define ALPHA 0.1f
#define KITER 10

// ---- CDNA5 async global->LDS copy (ASYNCcnt-tracked), per ISA 15.18.3 ----
__device__ __forceinline__ void async_b128(void* lds_dst, const void* gsrc) {
  asm volatile("global_load_async_to_lds_b128 %0, %1, off"
               :: "v"((unsigned)(unsigned long long)lds_dst), "v"(gsrc)
               : "memory");
}
#define WAIT_ASYNC(n) asm volatile("s_wait_asynccnt " #n ::: "memory")

__device__ __forceinline__ unsigned short f32_to_bf16(float f) {
  unsigned int u = __float_as_uint(f);
  u += 0x7FFFu + ((u >> 16) & 1u);          // round to nearest even
  return (unsigned short)(u >> 16);
}

__device__ __forceinline__ v16bf cat16(v8bf lo, v8bf hi) {
  return __builtin_shufflevector(lo, hi, 0,1,2,3,4,5,6,7,8,9,10,11,12,13,14,15);
}

// =======================================================================
// GEMM1: Hid[N,512](bf16) = relu(Xb[N,512](bf16) @ W0 + b0)
// W0 pre-transposed: W0bT[n][k] bf16. Tile 64x128, double-buffered async
// LDS copies with manual 2x unroll (constant buffer index -> stable acc regs).
// =======================================================================
__global__ __launch_bounds__(256)
void gemm1_bf16_wmma(const unsigned short* __restrict__ Xb,
                     const unsigned short* __restrict__ W0bT,  // [512 n][512 k]
                     const float* __restrict__ bias,           // [512]
                     unsigned short* __restrict__ Hout,        // [N,512] bf16
                     int nrows)
{
  __shared__ unsigned short lA[2][64][40];    // [m][k], 80B row stride
  __shared__ unsigned short lB[2][128][40];   // [n][k]
  const int t    = threadIdx.x;
  const int wid  = t >> 5, lane = t & 31;
  const int l15  = lane & 15, hi = lane >> 4;
  const int mi   = wid & 3, nh = wid >> 2;
  const int rowBase = blockIdx.x * 64;
  const int colBase = blockIdx.y * 128;

  v8f acc[4] = {{}, {}, {}, {}};

  auto issue = [&](int kk, int b) {
    {   // A tile 64x32 bf16 = 4KB : 1x16B per thread
      int row = t >> 2, koff = (t & 3) * 8;
      int gr = rowBase + row; if (gr >= nrows) gr = nrows - 1;  // clamp, stores guarded
      async_b128(&lA[b][row][koff], &Xb[(size_t)gr * HID + kk + koff]);
    }
#pragma unroll
    for (int j = 0; j < 2; ++j) {  // B tile 128x32 bf16 = 8KB : 2x16B per thread
      int c = t * 2 + j;           // 0..511
      int n = c >> 2, koff = (c & 3) * 8;
      async_b128(&lB[b][n][koff], &W0bT[(size_t)(colBase + n) * HID + kk + koff]);
    }
  };

  auto compute = [&](int b) {
    const int m = mi * 16 + l15;
    v16bf a = cat16(*(const v8bf*)&lA[b][m][hi * 16],
                    *(const v8bf*)&lA[b][m][hi * 16 + 8]);
    const int kb = hi * 16;
#pragma unroll
    for (int q = 0; q < 4; ++q) {
      int n = (nh * 4 + q) * 16 + l15;
      v16bf bb = cat16(*(const v8bf*)&lB[b][n][kb], *(const v8bf*)&lB[b][n][kb + 8]);
      acc[q] = __builtin_amdgcn_wmma_f32_16x16x32_bf16(false, a, false, bb,
                                                       (short)0, acc[q], false, false);
    }
  };

  const int NK = HID / 32;   // 16, even
  issue(0, 0);
#pragma unroll 1
  for (int i = 0; i < NK; i += 2) {
    // phase 0: consume buffer 0, prefetch into buffer 1
    issue((i + 1) * 32, 1);
    WAIT_ASYNC(3);
    __syncthreads();
    compute(0);
    __syncthreads();
    // phase 1: consume buffer 1, prefetch into buffer 0
    if (i + 2 < NK) { issue((i + 2) * 32, 0); WAIT_ASYNC(3); }
    else            { WAIT_ASYNC(0); }
    __syncthreads();
    compute(1);
    __syncthreads();
  }

#pragma unroll
  for (int v = 0; v < 8; ++v) {
    int gm = rowBase + mi * 16 + v + hi * 8;
    if (gm < nrows) {
#pragma unroll
      for (int q = 0; q < 4; ++q) {
        int gn = colBase + (nh * 4 + q) * 16 + l15;
        float x = acc[q][v] + bias[gn];
        x = x > 0.f ? x : 0.f;
        Hout[(size_t)gm * HID + gn] = f32_to_bf16(x);
      }
    }
  }
}

// =======================================================================
// GEMM2: h0[N,64](f32) = Hid[N,512](bf16) @ W1 + b1,  W1bT[n][k] bf16
// Tile 64x64, double-buffered async copies, manual 2x unroll.
// =======================================================================
__global__ __launch_bounds__(256)
void gemm2_bf16_wmma(const unsigned short* __restrict__ Hin,
                     const unsigned short* __restrict__ W1bT,  // [64 n][512 k]
                     const float* __restrict__ bias,           // [64]
                     float* __restrict__ Out,                  // [N,64] f32
                     int nrows)
{
  __shared__ unsigned short lA[2][64][40];
  __shared__ unsigned short lB[2][64][40];
  const int t    = threadIdx.x;
  const int wid  = t >> 5, lane = t & 31;
  const int l15  = lane & 15, hi = lane >> 4;
  const int mi   = wid & 3, nh = wid >> 2;
  const int ni0  = nh * 2, ni1 = nh * 2 + 1;
  const int rowBase = blockIdx.x * 64;

  v8f acc[2] = {{}, {}};

  auto issue = [&](int kk, int b) {
    {   // A tile 64x32 bf16
      int row = t >> 2, koff = (t & 3) * 8;
      int gr = rowBase + row; if (gr >= nrows) gr = nrows - 1;
      async_b128(&lA[b][row][koff], &Hin[(size_t)gr * HID + kk + koff]);
    }
    {   // B tile 64x32 bf16
      int n = t >> 2, koff = (t & 3) * 8;
      async_b128(&lB[b][n][koff], &W1bT[(size_t)n * HID + kk + koff]);
    }
  };

  auto compute = [&](int b) {
    const int m = mi * 16 + l15;
    v16bf a = cat16(*(const v8bf*)&lA[b][m][hi * 16],
                    *(const v8bf*)&lA[b][m][hi * 16 + 8]);
    const int kb = hi * 16;
    {
      int n = ni0 * 16 + l15;
      v16bf bb = cat16(*(const v8bf*)&lB[b][n][kb], *(const v8bf*)&lB[b][n][kb + 8]);
      acc[0] = __builtin_amdgcn_wmma_f32_16x16x32_bf16(false, a, false, bb,
                                                       (short)0, acc[0], false, false);
    }
    {
      int n = ni1 * 16 + l15;
      v16bf bb = cat16(*(const v8bf*)&lB[b][n][kb], *(const v8bf*)&lB[b][n][kb + 8]);
      acc[1] = __builtin_amdgcn_wmma_f32_16x16x32_bf16(false, a, false, bb,
                                                       (short)0, acc[1], false, false);
    }
  };

  const int NK = HID / 32;   // 16, even
  issue(0, 0);
#pragma unroll 1
  for (int i = 0; i < NK; i += 2) {
    issue((i + 1) * 32, 1);
    WAIT_ASYNC(2);
    __syncthreads();
    compute(0);
    __syncthreads();
    if (i + 2 < NK) { issue((i + 2) * 32, 0); WAIT_ASYNC(2); }
    else            { WAIT_ASYNC(0); }
    __syncthreads();
    compute(1);
    __syncthreads();
  }

#pragma unroll
  for (int v = 0; v < 8; ++v) {
    int gm = rowBase + mi * 16 + v + hi * 8;
    if (gm < nrows) {
      int gn0 = ni0 * 16 + l15;
      int gn1 = ni1 * 16 + l15;
      Out[(size_t)gm * CLS + gn0] = acc[0][v] + bias[gn0];
      Out[(size_t)gm * CLS + gn1] = acc[1][v] + bias[gn1];
    }
  }
}

// ---------------- conversion kernels ----------------
__global__ void cvt_f32_to_bf16_v4(const float* __restrict__ s,
                                   unsigned short* __restrict__ d, int n4) {
  int i = blockIdx.x * blockDim.x + threadIdx.x;
  if (i >= n4) return;
  float4 v = ((const float4*)s)[i];
  unsigned int w0 = (unsigned)f32_to_bf16(v.x) | ((unsigned)f32_to_bf16(v.y) << 16);
  unsigned int w1 = (unsigned)f32_to_bf16(v.z) | ((unsigned)f32_to_bf16(v.w) << 16);
  ((uint2*)d)[i] = make_uint2(w0, w1);
}

// d[n][k] = bf16(s[k][n]); k-dim is 512 (power of two) for both weights
__global__ void cvt_transpose_bf16(const float* __restrict__ s,
                                   unsigned short* __restrict__ d,
                                   int cols /*n-dim of s*/, int total) {
  int i = blockIdx.x * blockDim.x + threadIdx.x;
  if (i >= total) return;
  int k = i & (HID - 1);
  int n = i >> 9;
  d[i] = f32_to_bf16(s[(size_t)k * cols + n]);
}

// ---------------- graph kernels ----------------
__global__ void zero_f32(float* __restrict__ p, int n) {
  int i = blockIdx.x * blockDim.x + threadIdx.x;
  if (i < n) p[i] = 0.f;
}

__global__ void zero_f4(float4* __restrict__ p, int n4) {
  int i = blockIdx.x * blockDim.x + threadIdx.x;
  if (i < n4) p[i] = make_float4(0.f, 0.f, 0.f, 0.f);
}

__global__ void deg_accum(const int* __restrict__ dst, float* __restrict__ deg, int e) {
  int i = blockIdx.x * blockDim.x + threadIdx.x;
  if (i < e) atomicAdd(&deg[dst[i]], 1.0f);
}

__global__ void norm_finish(float* __restrict__ p, int n) {
  int i = blockIdx.x * blockDim.x + threadIdx.x;
  if (i < n) { float d = p[i]; p[i] = rsqrtf(d > 1.f ? d : 1.f); }
}

__global__ void init_hs(const float* __restrict__ h0, const float* __restrict__ nrm,
                        float* __restrict__ hs, int n16) {
  int t = blockIdx.x * blockDim.x + threadIdx.x;
  if (t >= n16) return;
  int node = t >> 4, c = (t & 15) * 4;
  float nv = nrm[node];
  float4 v = *(const float4*)&h0[(size_t)node * CLS + c];
  v.x *= nv; v.y *= nv; v.z *= nv; v.w *= nv;
  *(float4*)&hs[(size_t)node * CLS + c] = v;
}

__global__ void scatter_edges(const float* __restrict__ hs, float* __restrict__ agg,
                              const int* __restrict__ src, const int* __restrict__ dst,
                              int e16) {
  int t = blockIdx.x * blockDim.x + threadIdx.x;
  if (t >= e16) return;
  int e = t >> 4, c = (t & 15) * 4;
  int s = src[e], d = dst[e];
  float4 v = *(const float4*)&hs[(size_t)s * CLS + c];
  float* ap = &agg[(size_t)d * CLS + c];
  atomicAdd(ap + 0, v.x); atomicAdd(ap + 1, v.y);
  atomicAdd(ap + 2, v.z); atomicAdd(ap + 3, v.w);
}

__global__ void combine(const float* __restrict__ agg, const float* __restrict__ h0,
                        const float* __restrict__ nrm, float* __restrict__ hs,
                        float* __restrict__ hout, int n16) {
  int t = blockIdx.x * blockDim.x + threadIdx.x;
  if (t >= n16) return;
  int node = t >> 4, c = (t & 15) * 4;
  float nv = nrm[node];
  float4 a = *(const float4*)&agg[(size_t)node * CLS + c];
  float4 z = *(const float4*)&h0[(size_t)node * CLS + c];
  float4 h;
  h.x = (1.f - ALPHA) * a.x * nv + ALPHA * z.x;
  h.y = (1.f - ALPHA) * a.y * nv + ALPHA * z.y;
  h.z = (1.f - ALPHA) * a.z * nv + ALPHA * z.z;
  h.w = (1.f - ALPHA) * a.w * nv + ALPHA * z.w;
  *(float4*)&hout[(size_t)node * CLS + c] = h;
  float4 s = make_float4(h.x * nv, h.y * nv, h.z * nv, h.w * nv);
  *(float4*)&hs[(size_t)node * CLS + c] = s;
}

// ---------------- launcher ----------------
extern "C" void kernel_launch(void* const* d_in, const int* in_sizes, int n_in,
                              void* d_out, int out_size, void* d_ws, size_t ws_size,
                              hipStream_t stream) {
  (void)in_sizes; (void)n_in; (void)out_size; (void)ws_size;
  const float* features = (const float*)d_in[0];
  const float* W0 = (const float*)d_in[1];
  const float* b0 = (const float*)d_in[2];
  const float* W1 = (const float*)d_in[3];
  const float* b1 = (const float*)d_in[4];
  const int*   esrc = (const int*)d_in[5];
  const int*   edst = (const int*)d_in[6];
  float* out = (float*)d_out;

  char* ws = (char*)d_ws;
  size_t off = 0;
  auto carve = [&](size_t bytes) -> void* {
    void* p = ws + off;
    off = (off + bytes + 255) & ~(size_t)255;
    return p;
  };
  unsigned short* W0bT = (unsigned short*)carve((size_t)HID * HID * 2);
  unsigned short* W1bT = (unsigned short*)carve((size_t)CLS * HID * 2);
  float*          nrm  = (float*)carve((size_t)NN * 4);
  unsigned short* Xb   = (unsigned short*)carve((size_t)NN * HID * 2); // dead after GEMM1
  unsigned short* Hid  = (unsigned short*)carve((size_t)NN * HID * 2);
  float*          h0   = (float*)carve((size_t)NN * CLS * 4);
  // hs/agg alias the bf16-features region (only used after GEMM1 completes)
  float* hs  = (float*)Xb;
  float* agg = (float*)((char*)Xb + (((size_t)NN * CLS * 4 + 255) & ~(size_t)255));

  const int TB = 256;
  // weights: bf16 + transpose; features: bf16
  cvt_transpose_bf16<<<(HID * HID + TB - 1) / TB, TB, 0, stream>>>(W0, W0bT, HID, HID * HID);
  cvt_transpose_bf16<<<(HID * CLS + TB - 1) / TB, TB, 0, stream>>>(W1, W1bT, CLS, HID * CLS);
  {
    int n4 = NN * HID / 4;
    cvt_f32_to_bf16_v4<<<(n4 + TB - 1) / TB, TB, 0, stream>>>(features, Xb, n4);
  }
  // encoder
  dim3 g1((NN + 63) / 64, HID / 128);
  gemm1_bf16_wmma<<<g1, TB, 0, stream>>>(Xb, W0bT, b0, Hid, NN);
  dim3 g2((NN + 63) / 64, 1);
  gemm2_bf16_wmma<<<g2, TB, 0, stream>>>(Hid, W1bT, b1, h0, NN);

  // degree / norm
  zero_f32<<<(NN + TB - 1) / TB, TB, 0, stream>>>(nrm, NN);
  deg_accum<<<(NE + TB - 1) / TB, TB, 0, stream>>>(edst, nrm, NE);
  norm_finish<<<(NN + TB - 1) / TB, TB, 0, stream>>>(nrm, NN);

  // APPNP propagation (state is L2-resident: hs/agg/h0 + edge lists ~ 130MB < 192MB L2)
  const int n16 = NN * 16;
  const int e16 = NE * 16;
  init_hs<<<(n16 + TB - 1) / TB, TB, 0, stream>>>(h0, nrm, hs, n16);
  for (int k = 0; k < KITER; ++k) {
    zero_f4<<<(n16 + TB - 1) / TB, TB, 0, stream>>>((float4*)agg, n16);
    scatter_edges<<<(e16 + TB - 1) / TB, TB, 0, stream>>>(hs, agg, esrc, edst, e16);
    combine<<<(n16 + TB - 1) / TB, TB, 0, stream>>>(agg, h0, nrm, hs, out, n16);
  }
}
